// StreamingSVSModel_73117523247608
// MI455X (gfx1250) — matrix operand order repaired and checked by reading, not compiled, not run on hardware
//
#include <hip/hip_runtime.h>
#include <hip/hip_bf16.h>
#include <math.h>

// ---------------------------------------------------------------------------
// StreamingSVS forward for MI455X (gfx1250, wave32, WMMA).
//
// Roofline: ~0.61 TFLOP of GEMM (head blocks dominate) vs ~1 GB of traffic
// (~45us @ 23.3 TB/s) -> compute bound. We therefore run every matmul on
// v_wmma_f32_16x16x32_bf16 (f32 accumulate, bf16 inputs; 8x the K-depth of
// the f32 WMMA path). Residuals / LN / softmax / output stay f32.
// ---------------------------------------------------------------------------

typedef __bf16 v16bf  __attribute__((ext_vector_type(16)));
typedef __bf16 bhalf8 __attribute__((ext_vector_type(8)));
typedef float  v8f    __attribute__((ext_vector_type(8)));

#define TBM 128   // block tile M
#define TBN 64    // block tile N
#define TBK 32    // K step (one wmma K)
#define LDK 40    // padded LDS row stride (conflict-free: 80B rows, gcd(20,64)=4)

// ============================ WMMA GEMM ====================================
// out = act(A[M,K] @ W[N,K]^T + bias[N]) (+ resid[M,ldf]); A,W bf16 row-major.
// act: 0=none, 1=SiLU, 2=exact GELU.  outF (f32,ldf) and/or outB (bf16,ldb).
// Grid: (N/TBN, M/TBM), block 256 = 8 waves; wave (wm,wn) owns a 32x32 tile.
__global__ __launch_bounds__(256) void wmma_gemm_kernel(
    const __bf16* __restrict__ A, int lda,
    const __bf16* __restrict__ W, int ldw,
    const float* __restrict__ bias,
    const float* __restrict__ resid,
    float* __restrict__ outF, int ldf,
    __bf16* __restrict__ outB, int ldb,
    int K, int act)
{
  __shared__ __bf16 As[2][TBM][LDK];
  __shared__ __bf16 Ws[2][TBN][LDK];

  const int tid  = threadIdx.x;
  const int wave = tid >> 5;
  const int lane = tid & 31;
  const int wm   = wave & 3;        // 4 waves along M
  const int wn   = wave >> 2;       // 2 waves along N
  const int m0   = blockIdx.y * TBM;
  const int n0   = blockIdx.x * TBN;

  // cooperative tile loads: 32B (16 bf16) per thread per tile
  const int  arow  = tid >> 1;            // 0..127
  const int  ahalf = tid & 1;             // which 16-elem half of the 32-K row
  const int  wrow  = (tid & 127) >> 1;    // 0..63
  const bool doW   = tid < 128;

  const __bf16* aSrc = A + (size_t)(m0 + arow) * lda + ahalf * 16;
  const __bf16* wSrc = W + (size_t)(n0 + wrow) * ldw + ahalf * 16;

  { // preload K-tile 0 into buffer 0
    uint4 a0 = ((const uint4*)aSrc)[0];
    uint4 a1 = ((const uint4*)aSrc)[1];
    *((uint4*)&As[0][arow][ahalf * 16])     = a0;
    *((uint4*)&As[0][arow][ahalf * 16 + 8]) = a1;
    if (doW) {
      uint4 w0 = ((const uint4*)wSrc)[0];
      uint4 w1 = ((const uint4*)wSrc)[1];
      *((uint4*)&Ws[0][wrow][ahalf * 16])     = w0;
      *((uint4*)&Ws[0][wrow][ahalf * 16 + 8]) = w1;
    }
  }
  __syncthreads();

  v8f acc[2][2];
  #pragma unroll
  for (int i = 0; i < 2; ++i)
    #pragma unroll
    for (int j = 0; j < 2; ++j)
      #pragma unroll
      for (int r = 0; r < 8; ++r) acc[i][j][r] = 0.0f;

  const int lr = lane & 15;   // row (A) / col (B) within 16
  const int lh = lane >> 4;   // K-half selector per ISA layout
  const int KT = K / TBK;

  for (int kt = 0; kt < KT; ++kt) {
    const int  buf = kt & 1;
    const bool pre = (kt + 1 < KT);
    uint4 a0 = {0,0,0,0}, a1 = {0,0,0,0}, w0 = {0,0,0,0}, w1 = {0,0,0,0};
    if (pre) { // prefetch next K-tile into registers (overlaps with WMMA)
      const __bf16* ap = aSrc + (size_t)(kt + 1) * TBK;
      a0 = ((const uint4*)ap)[0];
      a1 = ((const uint4*)ap)[1];
      if (doW) {
        const __bf16* wp = wSrc + (size_t)(kt + 1) * TBK;
        w0 = ((const uint4*)wp)[0];
        w1 = ((const uint4*)wp)[1];
      }
    }

    // Build fragments from LDS per ISA 7.12.2 layouts (16-bit A 16x32 / B 32x16):
    //  A: lanes 0-15 hold K {0..7,16..23}, lanes 16-31 hold K {8..15,24..31}
    //  B: same K order as A: lanes 0-15 hold K {0..7,16..23} of col lr, lanes 16-31 hold K {8..15,24..31}
    v16bf af[2], wf[2];
    #pragma unroll
    for (int i = 0; i < 2; ++i) {
      const int row = wm * 32 + i * 16 + lr;
      const bhalf8 lo = *((const bhalf8*)&As[buf][row][lh * 8]);
      const bhalf8 hi = *((const bhalf8*)&As[buf][row][16 + lh * 8]);
      #pragma unroll
      for (int e = 0; e < 8; ++e) { af[i][e] = lo[e]; af[i][e + 8] = hi[e]; }
    }
    #pragma unroll
    for (int j = 0; j < 2; ++j) {
      const int col = wn * 32 + j * 16 + lr;
      const bhalf8 lo = *((const bhalf8*)&Ws[buf][col][lh * 8]);
      const bhalf8 hi = *((const bhalf8*)&Ws[buf][col][16 + lh * 8]);
      #pragma unroll
      for (int e = 0; e < 8; ++e) { wf[j][e] = lo[e]; wf[j][e + 8] = hi[e]; }
    }

    #pragma unroll
    for (int i = 0; i < 2; ++i)
      #pragma unroll
      for (int j = 0; j < 2; ++j)
        acc[i][j] = __builtin_amdgcn_wmma_f32_16x16x32_bf16(
            false, af[i], false, wf[j], (short)0, acc[i][j], false, false);

    __syncthreads();                 // everyone done reading LDS[buf]
    if (pre) {
      *((uint4*)&As[buf ^ 1][arow][ahalf * 16])     = a0;
      *((uint4*)&As[buf ^ 1][arow][ahalf * 16 + 8]) = a1;
      if (doW) {
        *((uint4*)&Ws[buf ^ 1][wrow][ahalf * 16])     = w0;
        *((uint4*)&Ws[buf ^ 1][wrow][ahalf * 16 + 8]) = w1;
      }
      __syncthreads();               // LDS[buf^1] ready for next iteration
    }
  }

  // Epilogue (C layout: lane L, VGPR r -> row r+8*(L>>4), col L&15)
  #pragma unroll
  for (int i = 0; i < 2; ++i) {
    const int rbase = m0 + wm * 32 + i * 16 + lh * 8;
    #pragma unroll
    for (int j = 0; j < 2; ++j) {
      const int col = n0 + wn * 32 + j * 16 + lr;
      const float bj = bias ? bias[col] : 0.0f;
      #pragma unroll
      for (int r = 0; r < 8; ++r) {
        float v = acc[i][j][r] + bj;
        if (act == 1)      v = v / (1.0f + expf(-v));                           // SiLU
        else if (act == 2) v = 0.5f * v * (1.0f + erff(v * 0.70710678118655f)); // GELU exact
        const size_t row = (size_t)(rbase + r);
        if (resid) v += resid[row * (size_t)ldf + col];
        if (outF)  outF[row * (size_t)ldf + col] = v;
        if (outB)  outB[row * (size_t)ldb + col] = (__bf16)v;
      }
    }
  }
}

// ===================== small support kernels ===============================

// f32 weight [N,Kin] -> bf16 [N,Kpad], zero-padding K
__global__ void k_w2bf(const float* __restrict__ w, __bf16* __restrict__ o,
                       int N, int Kin, int Kpad)
{
  int n = N * Kpad;
  for (int i = blockIdx.x * blockDim.x + threadIdx.x; i < n;
       i += gridDim.x * blockDim.x) {
    int r = i / Kpad, k = i - r * Kpad;
    o[i] = (k < Kin) ? (__bf16)w[(size_t)r * Kin + k] : (__bf16)0.0f;
  }
}

// gather [note|phoneme|slur|pg] embeddings -> cfeat bf16 [1024,320] (pad 304..319)
__global__ void k_cond_feat(const int* __restrict__ note, const int* __restrict__ phon,
                            const int* __restrict__ slur, const int* __restrict__ pg,
                            const float* __restrict__ note_e, const float* __restrict__ ph_e,
                            const float* __restrict__ sl_e, const float* __restrict__ pg_e,
                            __bf16* __restrict__ cfeat)
{
  const int row = blockIdx.x;     // b*T+t
  const int d   = threadIdx.x;    // 0..319
  int s = slur[row]; s = s < 0 ? 0 : (s > 1 ? 1 : s);
  float v = 0.0f;
  if      (d < 128) v = note_e[(size_t)note[row] * 128 + d];
  else if (d < 256) v = ph_e[(size_t)phon[row] * 128 + (d - 128)];
  else if (d < 272) v = sl_e[(size_t)s * 16 + (d - 256)];
  else if (d < 304) v = pg_e[(size_t)pg[row] * 32 + (d - 272)];
  cfeat[(size_t)row * 320 + d] = (__bf16)v;
}

// pooled[row,d] = mean over 32 codes of token_emb[code][d]   (bf16 out)
__global__ void k_pooled(const int* __restrict__ codes, const float* __restrict__ tok,
                         __bf16* __restrict__ pooled)
{
  const int row = blockIdx.x;
  const int d   = threadIdx.x;    // 0..255
  const int* c = codes + (size_t)row * 32;
  float s = 0.0f;
  #pragma unroll
  for (int j = 0; j < 32; ++j) s += tok[(size_t)c[j] * 256 + d];
  pooled[(size_t)row * 256 + d] = (__bf16)(s * (1.0f / 32.0f));
}

// xin[:,256:512] = shift-right(pr) with BOS at t==0
__global__ void k_shift(const __bf16* __restrict__ pr, const float* __restrict__ bos,
                        __bf16* __restrict__ xin)
{
  const int row = blockIdx.x;     // b*T+t
  const int d   = threadIdx.x;    // 0..255
  const int t   = row & 511;
  float v = (t == 0) ? bos[d] : (float)pr[(size_t)(row - 1) * 256 + d];
  xin[(size_t)row * 512 + 256 + d] = (__bf16)v;
}

// row-wise LayerNorm (f32 in) -> bf16 out; Dd multiple of blockDim
__global__ __launch_bounds__(256) void k_ln(const float* __restrict__ x,
                                            const float* __restrict__ g,
                                            const float* __restrict__ be,
                                            __bf16* __restrict__ o, int Dd)
{
  const int row = blockIdx.x;
  const float* xr = x + (size_t)row * Dd;
  float s = 0.0f, s2 = 0.0f;
  for (int d = threadIdx.x; d < Dd; d += 256) { float v = xr[d]; s += v; s2 += v * v; }
  #pragma unroll
  for (int off = 16; off > 0; off >>= 1) {
    s  += __shfl_xor(s,  off, 32);
    s2 += __shfl_xor(s2, off, 32);
  }
  __shared__ float rs[8], rs2[8];
  const int w = threadIdx.x >> 5, ln = threadIdx.x & 31;
  if (ln == 0) { rs[w] = s; rs2[w] = s2; }
  __syncthreads();
  if (w == 0) {
    float a = (ln < 8) ? rs[ln] : 0.0f;
    float b = (ln < 8) ? rs2[ln] : 0.0f;
    #pragma unroll
    for (int off = 4; off > 0; off >>= 1) {
      a += __shfl_xor(a, off, 32);
      b += __shfl_xor(b, off, 32);
    }
    if (ln == 0) { rs[0] = a; rs2[0] = b; }
  }
  __syncthreads();
  const float mean = rs[0] / Dd;
  const float var  = rs2[0] / Dd - mean * mean;
  const float rstd = rsqrtf(var + 1e-5f);
  for (int d = threadIdx.x; d < Dd; d += 256)
    o[(size_t)row * Dd + d] = (__bf16)((xr[d] - mean) * rstd * g[d] + be[d]);
}

// banded (WIN=64) causal attention, one 64-thread block per (b,h,t); f32 math.
// qkv layout [1024,1536]: q|k|v each 512 = 8 heads * 64
__global__ __launch_bounds__(64) void k_attn(const float* __restrict__ qkv,
                                             __bf16* __restrict__ o)
{
  const int t  = blockIdx.x;
  const int bh = blockIdx.y;
  const int b  = bh >> 3, h = bh & 7;
  const int j  = threadIdx.x;       // key slot / out dim
  __shared__ float sc[64];

  const int t0  = t - 63;
  const int key = t0 + j;
  const float* qp = qkv + ((size_t)(b * 512 + t)) * 1536 + h * 64;
  float s = -1e30f;
  if (key >= 0) {
    const float* kp = qkv + ((size_t)(b * 512 + key)) * 1536 + 512 + h * 64;
    float acc = 0.0f;
    #pragma unroll 8
    for (int d = 0; d < 64; ++d) acc += qp[d] * kp[d];
    s = acc * 0.125f;               // 1/sqrt(64)
  }
  sc[j] = s;
  __syncthreads();
  float mx = -1e30f;
  for (int i = 0; i < 64; ++i) mx = fmaxf(mx, sc[i]);
  float e = (key >= 0) ? expf(s - mx) : 0.0f;
  __syncthreads();
  sc[j] = e;
  __syncthreads();
  float sum = 0.0f;
  for (int i = 0; i < 64; ++i) sum += sc[i];
  float od = 0.0f;
  for (int i = 0; i < 64; ++i) {
    const int ky = t0 + i;
    if (ky < 0) continue;
    od += sc[i] * qkv[((size_t)(b * 512 + ky)) * 1536 + 1024 + h * 64 + j];
  }
  od /= sum;
  o[((size_t)(b * 512 + t)) * 512 + h * 64 + j] = (__bf16)od;
}

// H[m,d] = x[m>>5, d] + slot[m&31, d]
__global__ void k_head(const float* __restrict__ x, const float* __restrict__ slot,
                       float* __restrict__ H)
{
  const size_t n = (size_t)32768 * 512;
  for (size_t i = (size_t)blockIdx.x * blockDim.x + threadIdx.x; i < n;
       i += (size_t)gridDim.x * blockDim.x) {
    const int    d  = (int)(i & 511);
    const size_t m  = i >> 9;
    const int    fk = (int)(m & 31);
    const size_t bt = m >> 5;
    H[i] = x[bt * 512 + d] + slot[(size_t)fk * 512 + d];
  }
}

__global__ void k_f2b(const float* __restrict__ x, __bf16* __restrict__ o, size_t n)
{
  for (size_t i = (size_t)blockIdx.x * blockDim.x + threadIdx.x; i < n;
       i += (size_t)gridDim.x * blockDim.x)
    o[i] = (__bf16)x[i];
}

// ============================ launcher =====================================
// Param flattening assumption: harness flattens the params pytree like
// jax.tree_util (dict keys sorted alphabetically, lists in order):
//   d_in[0..4] = codes, note_id, phoneme_id, slur, phone_progress
//   5 + 12*l  : attn layer l : b1,b2,be1,be2,bo,bqkv,g1,g2,w1,w2,wo,wqkv
//   53 + 6*b  : block b      : b1,b2,be,g,w1,w2
//   77 bos, 78 chunk_b1, 79 chunk_b2, 80 chunk_w1, 81 chunk_w2, 82 cls_b,
//   83 cls_w, 84 cond_b1, 85 cond_b2, 86 cond_w1, 87 cond_w2, 88 note_emb,
//   89 pg_emb, 90 phoneme_emb, 91 prev_b1, 92 prev_b2, 93 prev_w1,
//   94 prev_w2, 95 slot, 96 slur_emb, 97 token_emb
extern "C" void kernel_launch(void* const* d_in, const int* in_sizes, int n_in,
                              void* d_out, int out_size, void* d_ws, size_t ws_size,
                              hipStream_t stream)
{
  (void)in_sizes; (void)n_in; (void)out_size; (void)ws_size;
  const int* codes   = (const int*)d_in[0];
  const int* note_id = (const int*)d_in[1];
  const int* phon_id = (const int*)d_in[2];
  const int* slur    = (const int*)d_in[3];
  const int* pg      = (const int*)d_in[4];
  auto P = [&](int i) { return (const float*)d_in[i]; };
  const int AL = 5, BL = 53;

  // ---- carve workspace (~320 MB needed) ----
  char* base = (char*)d_ws;
  size_t off = 0;
  auto carve = [&](size_t bytes) -> void* {
    void* p = base + off;
    off += (bytes + 255) & ~(size_t)255;
    return p;
  };
  // bf16 weights
  __bf16* w_cond1 = (__bf16*)carve((size_t)256 * 320 * 2);
  __bf16* w_cond2 = (__bf16*)carve((size_t)256 * 256 * 2);
  __bf16* w_prev1 = (__bf16*)carve((size_t)256 * 256 * 2);
  __bf16* w_prev2 = (__bf16*)carve((size_t)256 * 256 * 2);
  __bf16* w_chk1  = (__bf16*)carve((size_t)512 * 512 * 2);
  __bf16* w_chk2  = (__bf16*)carve((size_t)512 * 512 * 2);
  __bf16* w_cls   = (__bf16*)carve((size_t)1024 * 512 * 2);
  __bf16 *w_qkv[4], *w_o[4], *w_f1[4], *w_f2[4], *w_b1[4], *w_b2[4];
  for (int l = 0; l < 4; ++l) {
    w_qkv[l] = (__bf16*)carve((size_t)1536 * 512 * 2);
    w_o[l]   = (__bf16*)carve((size_t)512 * 512 * 2);
    w_f1[l]  = (__bf16*)carve((size_t)2048 * 512 * 2);
    w_f2[l]  = (__bf16*)carve((size_t)512 * 2048 * 2);
  }
  for (int b = 0; b < 4; ++b) {
    w_b1[b] = (__bf16*)carve((size_t)2048 * 512 * 2);
    w_b2[b] = (__bf16*)carve((size_t)512 * 2048 * 2);
  }
  // activations
  __bf16* cfeat  = (__bf16*)carve((size_t)1024 * 320 * 2);
  __bf16* pooled = (__bf16*)carve((size_t)1024 * 256 * 2);
  __bf16* c1     = (__bf16*)carve((size_t)1024 * 256 * 2);
  __bf16* xin    = (__bf16*)carve((size_t)1024 * 512 * 2);
  __bf16* pr1    = (__bf16*)carve((size_t)1024 * 256 * 2);
  __bf16* prt    = (__bf16*)carve((size_t)1024 * 256 * 2);
  __bf16* x1b    = (__bf16*)carve((size_t)1024 * 512 * 2);
  __bf16* hln    = (__bf16*)carve((size_t)1024 * 512 * 2);
  __bf16* attno  = (__bf16*)carve((size_t)1024 * 512 * 2);
  __bf16* ffh    = (__bf16*)carve((size_t)1024 * 2048 * 2);
  float*  x      = (float*)carve((size_t)1024 * 512 * 4);
  float*  qkv    = (float*)carve((size_t)1024 * 1536 * 4);
  float*  Hf     = (float*)carve((size_t)32768 * 512 * 4);
  __bf16* Hln    = (__bf16*)carve((size_t)32768 * 512 * 2);
  __bf16* Hhid   = (__bf16*)carve((size_t)32768 * 2048 * 2);

  auto conv = [&](const float* w, __bf16* o, int N, int Kin, int Kpad) {
    int n = N * Kpad;
    k_w2bf<<<dim3((n + 255) / 256), dim3(256), 0, stream>>>(w, o, N, Kin, Kpad);
  };
  auto gemm = [&](const __bf16* A, int lda, const __bf16* W, int ldw,
                  const float* bias, const float* resid,
                  float* outF, int ldf, __bf16* outB, int ldb,
                  int M, int N, int K, int act) {
    wmma_gemm_kernel<<<dim3(N / TBN, M / TBM), dim3(256), 0, stream>>>(
        A, lda, W, ldw, bias, resid, outF, ldf, outB, ldb, K, act);
  };

  // ---- weight conversion (per call; deterministic, tiny vs GEMM cost) ----
  conv(P(86), w_cond1, 256, 304, 320);
  conv(P(87), w_cond2, 256, 256, 256);
  conv(P(93), w_prev1, 256, 256, 256);
  conv(P(94), w_prev2, 256, 256, 256);
  conv(P(80), w_chk1, 512, 512, 512);
  conv(P(81), w_chk2, 512, 512, 512);
  conv(P(83), w_cls, 1024, 512, 512);
  for (int l = 0; l < 4; ++l) {
    const int p = AL + 12 * l;
    conv(P(p + 11), w_qkv[l], 1536, 512, 512);
    conv(P(p + 10), w_o[l],   512,  512, 512);
    conv(P(p + 8),  w_f1[l],  2048, 512, 512);
    conv(P(p + 9),  w_f2[l],  512, 2048, 2048);
  }
  for (int b = 0; b < 4; ++b) {
    const int p = BL + 6 * b;
    conv(P(p + 4), w_b1[b], 2048, 512, 512);
    conv(P(p + 5), w_b2[b], 512, 2048, 2048);
  }

  // ---- condition / prev encoders ----
  k_cond_feat<<<dim3(1024), dim3(320), 0, stream>>>(
      note_id, phon_id, slur, pg, P(88), P(90), P(96), P(89), cfeat);
  k_pooled<<<dim3(1024), dim3(256), 0, stream>>>(codes, P(97), pooled);

  gemm(cfeat, 320, w_cond1, 320, P(84), nullptr, nullptr, 0, c1, 256, 1024, 256, 320, 1);
  gemm(c1, 256, w_cond2, 256, P(85), nullptr, nullptr, 0, xin, 512, 1024, 256, 256, 0); // -> xin[:, :256]
  gemm(pooled, 256, w_prev1, 256, P(91), nullptr, nullptr, 0, pr1, 256, 1024, 256, 256, 1);
  gemm(pr1, 256, w_prev2, 256, P(92), nullptr, nullptr, 0, prt, 256, 1024, 256, 256, 0);
  k_shift<<<dim3(1024), dim3(256), 0, stream>>>(prt, P(77), xin);                        // -> xin[:, 256:]

  gemm(xin, 512, w_chk1, 512, P(78), nullptr, nullptr, 0, x1b, 512, 1024, 512, 512, 1);
  gemm(x1b, 512, w_chk2, 512, P(79), nullptr, x, 512, nullptr, 0, 1024, 512, 512, 0);

  // ---- attention trunk ----
  for (int l = 0; l < 4; ++l) {
    const int p = AL + 12 * l;
    k_ln<<<dim3(1024), dim3(256), 0, stream>>>(x, P(p + 6), P(p + 2), hln, 512);
    gemm(hln, 512, w_qkv[l], 512, P(p + 5), nullptr, qkv, 1536, nullptr, 0, 1024, 1536, 512, 0);
    k_attn<<<dim3(512, 16), dim3(64), 0, stream>>>(qkv, attno);
    gemm(attno, 512, w_o[l], 512, P(p + 4), x, x, 512, nullptr, 0, 1024, 512, 512, 0);
    k_ln<<<dim3(1024), dim3(256), 0, stream>>>(x, P(p + 7), P(p + 3), hln, 512);
    gemm(hln, 512, w_f1[l], 512, P(p + 0), nullptr, nullptr, 0, ffh, 2048, 1024, 2048, 512, 2);
    gemm(ffh, 2048, w_f2[l], 2048, P(p + 1), x, x, 512, nullptr, 0, 1024, 512, 2048, 0);
  }

  // ---- per-slot head (dominant cost) ----
  k_head<<<dim3(65536), dim3(256), 0, stream>>>(x, P(95), Hf);
  for (int b = 0; b < 4; ++b) {
    const int p = BL + 6 * b;
    k_ln<<<dim3(32768), dim3(256), 0, stream>>>(Hf, P(p + 3), P(p + 2), Hln, 512);
    gemm(Hln, 512, w_b1[b], 512, P(p + 0), nullptr, nullptr, 0, Hhid, 2048, 32768, 2048, 512, 2);
    gemm(Hhid, 2048, w_b2[b], 2048, P(p + 1), Hf, Hf, 512, nullptr, 0, 32768, 512, 2048, 0);
  }

  // ---- classifier -> d_out f32 [32768,1024] ----
  k_f2b<<<dim3(65536), dim3(256), 0, stream>>>(Hf, Hln, (size_t)32768 * 512);
  gemm(Hln, 512, w_cls, 512, P(82), nullptr, (float*)d_out, 1024, nullptr, 0, 32768, 1024, 512, 0);
}
